// SoftContrastiveLoss_77446850281979
// MI455X (gfx1250) — matrix operand order repaired and checked by reading, not compiled, not run on hardware
//
#include <hip/hip_runtime.h>
#include <math.h>

// Problem constants (from the reference)
#define B_SZ   4096
#define DEMB   256
#define DFEAT  512
#define NCB    64          // number of 64-wide column blocks = B_SZ/64
#define BK     32          // K chunk == WMMA bf16 K
#define BKP    (BK + 2)    // LDS row padding (68B stride -> good bank rotation)
#define TAU_MIN 0.001f
#define EPS_V   1e-8f

typedef __attribute__((ext_vector_type(16))) __bf16 v16bf;
typedef __attribute__((ext_vector_type(8)))  float  v8f;

// ---------------------------------------------------------------------------
// Kernel 1: zn = z / max(||z||,1e-12); emit bf16 hi/lo splits of zn and feat
// (x = hi + lo with ~16 mantissa bits total), and sq[i] = sum(feat[i]^2) in f32.
// One block (256 threads) per row.
// ---------------------------------------------------------------------------
__global__ void __launch_bounds__(256) prep_kernel(const float* __restrict__ z,
                                                   const float* __restrict__ feat,
                                                   __bf16* __restrict__ znh,
                                                   __bf16* __restrict__ znl,
                                                   __bf16* __restrict__ fh,
                                                   __bf16* __restrict__ fl,
                                                   float* __restrict__ sqf) {
  __shared__ float red[256];
  const int i = blockIdx.x;
  const int t = threadIdx.x;

  float zv = z[(size_t)i * DEMB + t];           // DEMB == blockDim.x == 256
  red[t] = zv * zv;
  __syncthreads();
  for (int s = 128; s > 0; s >>= 1) { if (t < s) red[t] += red[t + s]; __syncthreads(); }
  float inv = 1.0f / fmaxf(sqrtf(red[0]), 1e-12f);
  __syncthreads();
  float znv = zv * inv;
  __bf16 zh = (__bf16)znv;
  znh[(size_t)i * DEMB + t] = zh;
  znl[(size_t)i * DEMB + t] = (__bf16)(znv - (float)zh);

  float f0 = feat[(size_t)i * DFEAT + t];
  float f1 = feat[(size_t)i * DFEAT + 256 + t];
  __bf16 h0 = (__bf16)f0, h1 = (__bf16)f1;
  fh[(size_t)i * DFEAT + t]       = h0;
  fl[(size_t)i * DFEAT + t]       = (__bf16)(f0 - (float)h0);
  fh[(size_t)i * DFEAT + 256 + t] = h1;
  fl[(size_t)i * DFEAT + 256 + t] = (__bf16)(f1 - (float)h1);

  red[t] = f0 * f0 + f1 * f1;
  __syncthreads();
  for (int s = 128; s > 0; s >>= 1) { if (t < s) red[t] += red[t + s]; __syncthreads(); }
  if (t == 0) sqf[i] = red[0];
}

// ---------------------------------------------------------------------------
// bf16x3 split GEMM core (shared shape): block tile 128(M) x 64(N), 8 waves,
// each wave owns a 16x64 strip (4 C tiles).  Per K=32 chunk, per tile:
//   acc += Ahi*Bhi + Ahi*Blo + Alo*Bhi   (v_wmma_f32_16x16x32_bf16, f32 accum)
// Fragment layouts (16-bit A 16x32, B 32x16): both use contiguous k = 16h + s, not the ISA table order.
// ---------------------------------------------------------------------------
#define WMMA_BF16(A, Bv, C) \
  __builtin_amdgcn_wmma_f32_16x16x32_bf16(false, (A), false, (Bv), (short)0, (C), false, false)

// ---------------------------------------------------------------------------
// Kernel 2: dist[i][j] = sqrt(max(sq_i + sq_j - 2*feat_i.feat_j, 0)), diag=+inf
// ---------------------------------------------------------------------------
__global__ void __launch_bounds__(256) dist_kernel(const __bf16* __restrict__ fh,
                                                   const __bf16* __restrict__ fl,
                                                   const float* __restrict__ sqf,
                                                   float* __restrict__ dist) {
  __shared__ __bf16 Ah[128][BKP], Al[128][BKP];
  __shared__ __bf16 Bh[64][BKP],  Bl[64][BKP];

  const int row0 = blockIdx.y * 128;
  const int col0 = blockIdx.x * 64;
  const int t    = threadIdx.x;
  const int wave = t >> 5;
  const int lane = t & 31;
  const int half = lane >> 4;
  const int l16  = lane & 15;

  v8f acc0 = {0,0,0,0,0,0,0,0}, acc1 = acc0, acc2 = acc0, acc3 = acc0;

  const unsigned* gh = (const unsigned*)fh;   // 2 bf16 per dword
  const unsigned* gl = (const unsigned*)fl;

  for (int k0 = 0; k0 < DFEAT; k0 += BK) {
    const int ku0 = k0 >> 1;
    for (int idx = t; idx < 128 * (BK / 2); idx += 256) {
      int m = idx >> 4, ku = idx & 15;
      *(unsigned*)&Ah[m][ku * 2] = gh[(size_t)(row0 + m) * (DFEAT / 2) + ku0 + ku];
      *(unsigned*)&Al[m][ku * 2] = gl[(size_t)(row0 + m) * (DFEAT / 2) + ku0 + ku];
    }
    for (int idx = t; idx < 64 * (BK / 2); idx += 256) {
      int n = idx >> 4, ku = idx & 15;
      *(unsigned*)&Bh[n][ku * 2] = gh[(size_t)(col0 + n) * (DFEAT / 2) + ku0 + ku];
      *(unsigned*)&Bl[n][ku * 2] = gl[(size_t)(col0 + n) * (DFEAT / 2) + ku0 + ku];
    }
    __syncthreads();

    const int arow = wave * 16 + l16;
    v16bf ah, al;
#pragma unroll
    for (int e = 0; e < 8; e++) {               // A: v0-3 K=16h+e, v4-7 K=16h+8+e
      ah[e]     = Ah[arow][16 * half + e];
      ah[e + 8] = Ah[arow][16 * half + 8 + e];
      al[e]     = Al[arow][16 * half + e];
      al[e + 8] = Al[arow][16 * half + 8 + e];
    }
#pragma unroll
    for (int tt = 0; tt < 4; tt++) {
      const int brow = tt * 16 + l16;
      v16bf bh, bl;
#pragma unroll
      for (int e = 0; e < 16; e++) {            // B: lanes 0-15 K=0-15, 16-31 K=16-31
        bh[e] = Bh[brow][16 * half + e];
        bl[e] = Bl[brow][16 * half + e];
      }
      v8f a = (tt == 0 ? acc0 : tt == 1 ? acc1 : tt == 2 ? acc2 : acc3);
      a = WMMA_BF16(ah, bh, a);
      a = WMMA_BF16(ah, bl, a);
      a = WMMA_BF16(al, bh, a);
      if (tt == 0) acc0 = a; else if (tt == 1) acc1 = a; else if (tt == 2) acc2 = a; else acc3 = a;
    }
    __syncthreads();
  }

  // C/D layout: VGPR r, lanes 0-15 -> (M=r, N=lane); lanes 16-31 -> (M=r+8, N=lane-16)
  const int ibase = row0 + wave * 16;
#pragma unroll
  for (int r = 0; r < 8; r++) {
    const int i = ibase + r + half * 8;
    const float si = sqf[i];
#pragma unroll
    for (int tt = 0; tt < 4; tt++) {
      const int j = col0 + tt * 16 + l16;
      float g = (tt == 0 ? acc0[r] : tt == 1 ? acc1[r] : tt == 2 ? acc2[r] : acc3[r]);
      float d2 = fmaxf(si + sqf[j] - 2.0f * g, 0.0f);
      float d = (i == j) ? __builtin_inff() : sqrtf(d2);
      dist[(size_t)i * B_SZ + j] = d;
    }
  }
}

// ---------------------------------------------------------------------------
// Kernel 3: per-row exact median (k = 2047-th smallest, 0-indexed) via 4-pass
// radix select on float bits (all values >= 0, +inf sorts last); then
// tau = max(med, 0.001), row min (softmax shift), W, scale, sum(p).
// One block (256 threads) per row.
// ---------------------------------------------------------------------------
__global__ void __launch_bounds__(256) median_kernel(const float* __restrict__ dist,
                                                     float* __restrict__ invtau,
                                                     float* __restrict__ moff,
                                                     float* __restrict__ scalep,
                                                     float* __restrict__ psum) {
  __shared__ unsigned rowb[B_SZ];
  __shared__ unsigned bins[256];
  __shared__ unsigned sh_prefix;
  __shared__ int sh_k;
  __shared__ float fred[256];

  const int i = blockIdx.x;
  const int t = threadIdx.x;
  const float* row = dist + (size_t)i * B_SZ;

  for (int e = t; e < B_SZ; e += 256) rowb[e] = __float_as_uint(row[e]);
  __syncthreads();

  // row min (diag = inf never wins)
  float mn = 3.0e38f;
  for (int e = t; e < B_SZ; e += 256) mn = fminf(mn, __uint_as_float(rowb[e]));
  fred[t] = mn; __syncthreads();
  for (int s = 128; s > 0; s >>= 1) { if (t < s) fred[t] = fminf(fred[t], fred[t + s]); __syncthreads(); }
  const float dmin = fred[0];
  __syncthreads();

  if (t == 0) { sh_prefix = 0u; sh_k = (B_SZ - 2) / 2; }   // 2047
  __syncthreads();

  for (int pass = 0; pass < 4; pass++) {
    const int shift = 24 - 8 * pass;
    bins[t] = 0u;
    __syncthreads();
    const unsigned prefix = sh_prefix;
    for (int e = t; e < B_SZ; e += 256) {
      unsigned u = rowb[e];
      if (pass == 0 || ((u ^ prefix) >> (shift + 8)) == 0u)   // short-circuits pass 0
        atomicAdd(&bins[(u >> shift) & 255u], 1u);
    }
    __syncthreads();
    if (t == 0) {
      int k = sh_k; unsigned cum = 0;
      for (int b = 0; b < 256; b++) {
        unsigned c = bins[b];
        if ((unsigned)k < cum + c) { sh_prefix = prefix | ((unsigned)b << shift); sh_k = k - (int)cum; break; }
        cum += c;
      }
    }
    __syncthreads();
  }

  const float med  = __uint_as_float(sh_prefix);
  const float tau  = fmaxf(med, TAU_MIN);
  const float itau = 1.0f / tau;

  // W = sum_j exp((dmin - d_ij)/tau)  (diag: d=inf -> contributes 0)
  float wsum = 0.0f;
  for (int e = t; e < B_SZ; e += 256) {
    float d = __uint_as_float(rowb[e]);
    wsum += __expf((dmin - d) * itau);
  }
  fred[t] = wsum; __syncthreads();
  for (int s = 128; s > 0; s >>= 1) { if (t < s) fred[t] += fred[t + s]; __syncthreads(); }
  if (t == 0) {
    float W  = fred[0];
    float sc = (W > EPS_V) ? (1.0f / (W + EPS_V)) : 1.0f;
    invtau[i] = itau;
    moff[i]   = dmin * itau;     // p = exp(moff - d*itau) * sc
    scalep[i] = sc;
    psum[i]   = W * sc;          // sum_j p_ij
  }
}

// ---------------------------------------------------------------------------
// Kernel 4: bf16x3 WMMA GEMM on zn (K=256) fused with loss epilogue.
// Per element: s = 10 * (zn_i . zn_j); E += exp(s); T += p_ij * s.
// Per-(row, col-block) partials written without atomics (deterministic).
// ---------------------------------------------------------------------------
__global__ void __launch_bounds__(256) embloss_kernel(const __bf16* __restrict__ znh,
                                                      const __bf16* __restrict__ znl,
                                                      const float* __restrict__ dist,
                                                      const float* __restrict__ invtau,
                                                      const float* __restrict__ moff,
                                                      const float* __restrict__ scalep,
                                                      float* __restrict__ Epart,
                                                      float* __restrict__ Tpart) {
  __shared__ __bf16 Ah[128][BKP], Al[128][BKP];
  __shared__ __bf16 Bh[64][BKP],  Bl[64][BKP];

  const int row0 = blockIdx.y * 128;
  const int col0 = blockIdx.x * 64;
  const int t    = threadIdx.x;
  const int wave = t >> 5;
  const int lane = t & 31;
  const int half = lane >> 4;
  const int l16  = lane & 15;

  v8f acc0 = {0,0,0,0,0,0,0,0}, acc1 = acc0, acc2 = acc0, acc3 = acc0;

  const unsigned* gh = (const unsigned*)znh;
  const unsigned* gl = (const unsigned*)znl;

  for (int k0 = 0; k0 < DEMB; k0 += BK) {
    const int ku0 = k0 >> 1;
    for (int idx = t; idx < 128 * (BK / 2); idx += 256) {
      int m = idx >> 4, ku = idx & 15;
      *(unsigned*)&Ah[m][ku * 2] = gh[(size_t)(row0 + m) * (DEMB / 2) + ku0 + ku];
      *(unsigned*)&Al[m][ku * 2] = gl[(size_t)(row0 + m) * (DEMB / 2) + ku0 + ku];
    }
    for (int idx = t; idx < 64 * (BK / 2); idx += 256) {
      int n = idx >> 4, ku = idx & 15;
      *(unsigned*)&Bh[n][ku * 2] = gh[(size_t)(col0 + n) * (DEMB / 2) + ku0 + ku];
      *(unsigned*)&Bl[n][ku * 2] = gl[(size_t)(col0 + n) * (DEMB / 2) + ku0 + ku];
    }
    __syncthreads();

    const int arow = wave * 16 + l16;
    v16bf ah, al;
#pragma unroll
    for (int e = 0; e < 8; e++) {
      ah[e]     = Ah[arow][16 * half + e];
      ah[e + 8] = Ah[arow][16 * half + 8 + e];
      al[e]     = Al[arow][16 * half + e];
      al[e + 8] = Al[arow][16 * half + 8 + e];
    }
#pragma unroll
    for (int tt = 0; tt < 4; tt++) {
      const int brow = tt * 16 + l16;
      v16bf bh, bl;
#pragma unroll
      for (int e = 0; e < 16; e++) {
        bh[e] = Bh[brow][16 * half + e];
        bl[e] = Bl[brow][16 * half + e];
      }
      v8f a = (tt == 0 ? acc0 : tt == 1 ? acc1 : tt == 2 ? acc2 : acc3);
      a = WMMA_BF16(ah, bh, a);
      a = WMMA_BF16(ah, bl, a);
      a = WMMA_BF16(al, bh, a);
      if (tt == 0) acc0 = a; else if (tt == 1) acc1 = a; else if (tt == 2) acc2 = a; else acc3 = a;
    }
    __syncthreads();
  }

  const int ibase = row0 + wave * 16;
#pragma unroll
  for (int r = 0; r < 8; r++) {
    const int i  = ibase + r + half * 8;
    const float it = invtau[i];
    const float mo = moff[i];
    const float sc = scalep[i];
    float vE = 0.0f, vT = 0.0f;
#pragma unroll
    for (int tt = 0; tt < 4; tt++) {
      const int j = col0 + tt * 16 + l16;
      float g = (tt == 0 ? acc0[r] : tt == 1 ? acc1[r] : tt == 2 ? acc2[r] : acc3[r]);
      float s = g * 10.0f;                 // / TEMPERATURE
      vE += __expf(s);
      if (i != j) {
        float d = dist[(size_t)i * B_SZ + j];
        float p = __expf(mo - d * it) * sc;
        vT += p * s;
      }
    }
    // reduce across the 16 lanes holding this row's columns
    for (int m = 1; m < 16; m <<= 1) {
      vE += __shfl_xor(vE, m, 32);
      vT += __shfl_xor(vT, m, 32);
    }
    if (l16 == 0) {
      Epart[(size_t)i * NCB + blockIdx.x] = vE;
      Tpart[(size_t)i * NCB + blockIdx.x] = vT;
    }
  }
}

// ---------------------------------------------------------------------------
// Kernel 5: loss = mean_i [ log(E_i) * psum_i - T_i ],  nan->0, +-inf->1e4
// ---------------------------------------------------------------------------
__global__ void __launch_bounds__(256) finalize_kernel(const float* __restrict__ Epart,
                                                       const float* __restrict__ Tpart,
                                                       const float* __restrict__ psum,
                                                       float* __restrict__ out) {
  __shared__ float red[256];
  const int t = threadIdx.x;
  float acc = 0.0f;
  for (int i = t; i < B_SZ; i += 256) {
    float E = 0.0f, T = 0.0f;
    for (int cb = 0; cb < NCB; cb++) {
      E += Epart[(size_t)i * NCB + cb];
      T += Tpart[(size_t)i * NCB + cb];
    }
    acc += logf(E) * psum[i] - T;
  }
  red[t] = acc; __syncthreads();
  for (int s = 128; s > 0; s >>= 1) { if (t < s) red[t] += red[t + s]; __syncthreads(); }
  if (t == 0) {
    float l = red[0] / (float)B_SZ;
    if (l != l) l = 0.0f;                                   // nan -> 0
    else if (l > 3.0e38f || l < -3.0e38f) l = 1e4f;         // +-inf -> 1e4
    out[0] = l;
  }
}

// ---------------------------------------------------------------------------
extern "C" void kernel_launch(void* const* d_in, const int* in_sizes, int n_in,
                              void* d_out, int out_size, void* d_ws, size_t ws_size,
                              hipStream_t stream) {
  (void)in_sizes; (void)n_in; (void)out_size; (void)ws_size;
  const float* z    = (const float*)d_in[0];
  const float* feat = (const float*)d_in[1];

  float* dist   = (float*)d_ws;                         // B*B            (64 MB)
  float* Epart  = dist   + (size_t)B_SZ * B_SZ;         // B*NCB
  float* Tpart  = Epart  + (size_t)B_SZ * NCB;          // B*NCB
  float* sqf    = Tpart  + (size_t)B_SZ * NCB;          // B
  float* invtau = sqf    + B_SZ;                        // B
  float* moff   = invtau + B_SZ;                        // B
  float* scalep = moff   + B_SZ;                        // B
  float* psumv  = scalep + B_SZ;                        // B
  __bf16* fh  = (__bf16*)(psumv + B_SZ);                // B*DFEAT bf16 (4 MB)
  __bf16* fl  = fh  + (size_t)B_SZ * DFEAT;             // B*DFEAT bf16
  __bf16* znh = fl  + (size_t)B_SZ * DFEAT;             // B*DEMB bf16 (2 MB)
  __bf16* znl = znh + (size_t)B_SZ * DEMB;              // B*DEMB bf16

  prep_kernel<<<B_SZ, 256, 0, stream>>>(z, feat, znh, znl, fh, fl, sqf);

  dim3 gtile(B_SZ / 64, B_SZ / 128);   // (64, 32)
  dist_kernel<<<gtile, 256, 0, stream>>>(fh, fl, sqf, dist);

  median_kernel<<<B_SZ, 256, 0, stream>>>(dist, invtau, moff, scalep, psumv);

  embloss_kernel<<<gtile, 256, 0, stream>>>(znh, znl, dist, invtau, moff, scalep, Epart, Tpart);

  finalize_kernel<<<1, 256, 0, stream>>>(Epart, Tpart, psumv, (float*)d_out);
}
